// InferSent_10196252360948
// MI455X (gfx1250) — matrix operand order repaired and checked by reading, compiled, not run on hardware
//
#include <hip/hip_runtime.h>
#include <hip/hip_bf16.h>
#include <math.h>

// ---------------------------------------------------------------------------
// Problem constants (fixed by the reference harness)
// ---------------------------------------------------------------------------
#define T_LEN 8192
#define D_IN  512
#define H_DIM 2048
#define S_SEG 512
#define SEG_L 16           // T/S : hidden state resets every 16 tokens
#define TH    (3 * H_DIM)  // 6144
#define EPS_W 1e-4f

// ---------------------------------------------------------------------------
// bf16 helpers (round-to-nearest-even pack, cheap unpack)
// ---------------------------------------------------------------------------
__device__ __forceinline__ unsigned short f2bf(float f) {
  unsigned int u = __builtin_bit_cast(unsigned int, f);
  unsigned int r = (u + 0x7FFFu + ((u >> 16) & 1u)) >> 16;
  return (unsigned short)r;
}
__device__ __forceinline__ float bf2f(unsigned short s) {
  unsigned int u = ((unsigned int)s) << 16;
  return __builtin_bit_cast(float, u);
}
__device__ __forceinline__ float sigm(float x) { return 1.0f / (1.0f + expf(-x)); }

// ---------------------------------------------------------------------------
// WMMA types (CDNA5 / gfx1250: V_WMMA_F32_16X16X32_BF16)
// ---------------------------------------------------------------------------
typedef __attribute__((ext_vector_type(16))) __bf16 bf16x16;
typedef __attribute__((ext_vector_type(8)))  float  floatx8;

union Frag16 { bf16x16 v; unsigned int u[8]; };

// ---------------------------------------------------------------------------
// CDNA5 async global->LDS copy (ASYNCcnt-tracked; no VGPR staging, no
// per-transfer loadcnt wait).  LDS byte address = low 32 bits of the generic
// shared pointer (flat aperture keeps the LDS offset in addr[31:0], ISA 10.2).
// ---------------------------------------------------------------------------
__device__ __forceinline__ void async_copy_b128(unsigned lds_byte,
                                                const unsigned short* g) {
  asm volatile("global_load_async_to_lds_b128 %0, %1, off"
               :: "v"(lds_byte), "v"(g)
               : "memory");
}
__device__ __forceinline__ void wait_async_le4() {
  asm volatile("s_wait_asynccnt 0x4" ::: "memory");
}
__device__ __forceinline__ void wait_async_le0() {
  asm volatile("s_wait_asynccnt 0x0" ::: "memory");
}

// ---------------------------------------------------------------------------
// Generic bf16 WMMA GEMM:  C(MxN) = act( A(MxK) * B(NxK)^T + bias )
//   A rows may be gathered:  physical A row = logical_row * rowmul + rowoff
//   (used to pull token rows {16*s + k} for the batched-recurrence step GEMM)
// Block tile 128x128, K-chunk 32, 8 waves (wave32), wave tile 32x64 = 2x4 WMMA.
// Double-buffered LDS filled by GLOBAL_LOAD_ASYNC_TO_LDS_B128.
// Requires: M%128==0, N%128==0, K%32==0 (true for all shapes here).
// ---------------------------------------------------------------------------
#define BM 128
#define BN 128
#define BK 32
#define LDSS 40  // padded bf16 row stride (80B): bank-conflict-free column reads

__global__ __launch_bounds__(256) void gemm_bf16_wmma(
    const unsigned short* __restrict__ A, int lda, int rowmul, int rowoff,
    const unsigned short* __restrict__ B, int ldb,
    const float* __restrict__ bias, int act,
    float* __restrict__ Cf, unsigned short* __restrict__ Cbf, int ldc,
    int K)
{
  __shared__ __align__(16) unsigned short smem[2 * (BM + BN) * LDSS];

  const int tid  = threadIdx.x;
  const int wave = tid >> 5;
  const int lane = tid & 31;
  const int lrow = lane & 15;
  const int half = lane >> 4;

  const int bm = blockIdx.y * BM;
  const int bn = blockIdx.x * BN;
  const int wm = (wave & 3) * 32;   // wave M offset inside block tile
  const int wn = (wave >> 2) * 64;  // wave N offset inside block tile

  // Per-lane K offsets inside a K=32 chunk (ISA 7.12.2 bf16 layouts)
  int ka[8], kb[8];
#pragma unroll
  for (int v = 0; v < 8; ++v) {
    ka[v] = 16 * (v >> 2) + 8 * half + 2 * (v & 3);  // A 16x32 layout
    kb[v] = 16 * (v >> 2) + 8 * half + 2 * (v & 3);  // B 32x16 layout
  }

  floatx8 zero = {0.f, 0.f, 0.f, 0.f, 0.f, 0.f, 0.f, 0.f};
  floatx8 acc[2][4];
#pragma unroll
  for (int r = 0; r < 2; ++r)
#pragma unroll
    for (int c = 0; c < 4; ++c) acc[r][c] = zero;

  // Loader mapping: 4 threads x 16B cover one 64B row of 32 bf16;
  // 2 passes x {A,B} = 4 async B128 copies per thread per K-chunk.
  const int ldr   = tid >> 2;        // 0..63
  const int ldc16 = (tid & 3) * 8;   // bf16 column: 0,8,16,24

  const size_t arow0 = (size_t)((bm + ldr)      * rowmul + rowoff) * lda + ldc16;
  const size_t arow1 = (size_t)((bm + ldr + 64) * rowmul + rowoff) * lda + ldc16;
  const size_t brow0 = (size_t)(bn + ldr)      * ldb + ldc16;
  const size_t brow1 = (size_t)(bn + ldr + 64) * ldb + ldc16;

  const unsigned smem_base = (unsigned)(unsigned long long)&smem[0];
  const unsigned bufstride = (BM + BN) * LDSS * 2;  // bytes per buffer
  const unsigned a_off0 = (unsigned)(((ldr)      * LDSS + ldc16) * 2);
  const unsigned a_off1 = (unsigned)(((ldr + 64) * LDSS + ldc16) * 2);
  const unsigned b_off0 = (unsigned)(((BM + ldr)      * LDSS + ldc16) * 2);
  const unsigned b_off1 = (unsigned)(((BM + ldr + 64) * LDSS + ldc16) * 2);

  const int nch = K / BK;

  // prologue: chunk 0 -> buffer 0
  {
    const unsigned base = smem_base;
    async_copy_b128(base + a_off0, A + arow0);
    async_copy_b128(base + a_off1, A + arow1);
    async_copy_b128(base + b_off0, B + brow0);
    async_copy_b128(base + b_off1, B + brow1);
  }

  for (int c = 0; c < nch; ++c) {
    const int cur = c & 1;
    if (c + 1 < nch) {
      // issue next chunk into the other buffer, then wait for current chunk
      const int kc = (c + 1) * BK;
      const unsigned base = smem_base + (unsigned)(cur ^ 1) * bufstride;
      async_copy_b128(base + a_off0, A + arow0 + kc);
      async_copy_b128(base + a_off1, A + arow1 + kc);
      async_copy_b128(base + b_off0, B + brow0 + kc);
      async_copy_b128(base + b_off1, B + brow1 + kc);
      wait_async_le4();   // async loads complete in order: oldest 4 == chunk c
    } else {
      wait_async_le0();
    }
    __syncthreads();      // all waves' async fills for chunk c visible

    const unsigned short* As = smem + (size_t)cur * (BM + BN) * LDSS;
    const unsigned short* Bs = As + BM * LDSS;

    Frag16 af[2], bg[4];
#pragma unroll
    for (int r = 0; r < 2; ++r) {
      const int row = wm + r * 16 + lrow;
#pragma unroll
      for (int v = 0; v < 8; ++v)
        af[r].u[v] = *(const unsigned int*)(As + row * LDSS + ka[v]);
    }
#pragma unroll
    for (int cc = 0; cc < 4; ++cc) {
      const int row = wn + cc * 16 + lrow;
#pragma unroll
      for (int v = 0; v < 8; ++v)
        bg[cc].u[v] = *(const unsigned int*)(Bs + row * LDSS + kb[v]);
    }
#pragma unroll
    for (int r = 0; r < 2; ++r)
#pragma unroll
      for (int cc = 0; cc < 4; ++cc)
        acc[r][cc] = __builtin_amdgcn_wmma_f32_16x16x32_bf16(
            false, af[r].v, false, bg[cc].v, (short)0, acc[r][cc], false, false);

    __syncthreads();      // buffer cur may be overwritten next iteration
  }

  // Write-out: C/D layout lane->N (=lrow), VGPR v -> M = v + 8*half
#pragma unroll
  for (int r = 0; r < 2; ++r) {
#pragma unroll
    for (int c = 0; c < 4; ++c) {
      const int ng = bn + wn + c * 16 + lrow;
      float bv = bias ? bias[ng] : 0.0f;
#pragma unroll
      for (int v = 0; v < 8; ++v) {
        const int mg = bm + wm + r * 16 + v + 8 * half;
        float val = acc[r][c][v] + bv;
        if (act) val = tanhf(val);
        const size_t idx = (size_t)mg * ldc + ng;
        if (Cf)  Cf[idx]  = val;
        if (Cbf) Cbf[idx] = f2bf(val);
      }
    }
  }
}

// ---------------------------------------------------------------------------
// fp32 -> bf16 cast (one pass over weights / input at launch)
// ---------------------------------------------------------------------------
__global__ void cast_bf16_kernel(const float* __restrict__ src,
                                 unsigned short* __restrict__ dst, int n) {
  int i = blockIdx.x * 256 + threadIdx.x;
  if (i < n) dst[i] = f2bf(src[i]);
}

__global__ void zero_state_kernel(float* __restrict__ hf,
                                  unsigned short* __restrict__ hbf, int n) {
  int i = blockIdx.x * 256 + threadIdx.x;
  if (i < n) { hf[i] = 0.0f; if (hbf) hbf[i] = 0; }
}

// ---------------------------------------------------------------------------
// Batched lower-GRU elementwise update for step k (512 segments in parallel)
// ---------------------------------------------------------------------------
__global__ void gru1_update_kernel(const float* __restrict__ Gi,
                                   const float* __restrict__ Gh,
                                   float* __restrict__ Hf,
                                   unsigned short* __restrict__ Hbf,
                                   unsigned short* __restrict__ outs_bf, int k) {
  int idx = blockIdx.x * 256 + threadIdx.x;      // [0, S*H)
  int s = idx >> 11, j = idx & (H_DIM - 1);
  size_t g = (size_t)s * TH;
  float r = sigm(Gi[g + j] + Gh[g + j]);
  float z = sigm(Gi[g + H_DIM + j] + Gh[g + H_DIM + j]);
  float n = tanhf(Gi[g + 2 * H_DIM + j] + r * Gh[g + 2 * H_DIM + j]);
  float hn = (1.0f - z) * n + z * Hf[idx];
  Hf[idx] = hn;
  Hbf[idx] = f2bf(hn);
  outs_bf[(size_t)(SEG_L * s + k) * H_DIM + j] = f2bf(hn);
}

// ---------------------------------------------------------------------------
// Row dot with vector + degenerate softmax  w = e/(e+eps)
// ---------------------------------------------------------------------------
__global__ __launch_bounds__(256) void row_dot_bf16_kernel(
    const unsigned short* __restrict__ X, const float* __restrict__ w,
    float* __restrict__ out, int K) {
  __shared__ float red[256];
  const unsigned short* xr = X + (size_t)blockIdx.x * K;
  float s = 0.0f;
  for (int i = threadIdx.x; i < K; i += 256) s += bf2f(xr[i]) * w[i];
  red[threadIdx.x] = s;
  __syncthreads();
  for (int off = 128; off > 0; off >>= 1) {
    if (threadIdx.x < off) red[threadIdx.x] += red[threadIdx.x + off];
    __syncthreads();
  }
  if (threadIdx.x == 0) { float e = expf(red[0]); out[blockIdx.x] = e / (e + EPS_W); }
}

__global__ __launch_bounds__(256) void row_dot_f32_kernel(
    const float* __restrict__ X, const float* __restrict__ w,
    float* __restrict__ out, int K) {
  __shared__ float red[256];
  const float* xr = X + (size_t)blockIdx.x * K;
  float s = 0.0f;
  for (int i = threadIdx.x; i < K; i += 256) s += xr[i] * w[i];
  red[threadIdx.x] = s;
  __syncthreads();
  for (int off = 128; off > 0; off >>= 1) {
    if (threadIdx.x < off) red[threadIdx.x] += red[threadIdx.x + off];
    __syncthreads();
  }
  if (threadIdx.x == 0) { float e = expf(red[0]); out[blockIdx.x] = e / (e + EPS_W); }
}

// ---------------------------------------------------------------------------
// Per-segment max of w[t]*emb_h[t][j] over the 16 tokens of each segment
// ---------------------------------------------------------------------------
__global__ void seg_max_kernel(const unsigned short* __restrict__ embh,
                               const float* __restrict__ w,
                               unsigned short* __restrict__ seg_bf) {
  int idx = blockIdx.x * 256 + threadIdx.x;      // [0, S*H)
  int s = idx >> 11, j = idx & (H_DIM - 1);
  float m = -3.4e38f;
#pragma unroll
  for (int k = 0; k < SEG_L; ++k) {
    int t = SEG_L * s + k;
    m = fmaxf(m, w[t] * bf2f(embh[(size_t)t * H_DIM + j]));
  }
  seg_bf[idx] = f2bf(m);
}

// ---------------------------------------------------------------------------
// Upper-GRU step: gh = Wh2 @ h + bh2 (6144x2048 GEMV; Wh2 is L2-resident)
// ---------------------------------------------------------------------------
__global__ __launch_bounds__(256) void gemv_step_kernel(
    const unsigned short* __restrict__ Wh, const float* __restrict__ bh,
    const float* __restrict__ h, float* __restrict__ gh) {
  int row = blockIdx.x * 8 + (threadIdx.x >> 5);
  int lane = threadIdx.x & 31;
  const unsigned short* wr = Wh + (size_t)row * H_DIM;
  float s = 0.0f;
  for (int i = lane; i < H_DIM; i += 32) s += bf2f(wr[i]) * h[i];
  for (int off = 16; off > 0; off >>= 1) s += __shfl_down(s, off, 32);
  if (lane == 0) gh[row] = s + bh[row];
}

__global__ void gru2_update_kernel(const float* __restrict__ gi,
                                   const float* __restrict__ gh,
                                   float* __restrict__ h2,
                                   unsigned short* __restrict__ out_bf) {
  int j = blockIdx.x * 256 + threadIdx.x;        // [0, H)
  float r = sigm(gi[j] + gh[j]);
  float z = sigm(gi[H_DIM + j] + gh[H_DIM + j]);
  float n = tanhf(gi[2 * H_DIM + j] + r * gh[2 * H_DIM + j]);
  float hn = (1.0f - z) * n + z * h2[j];
  h2[j] = hn;
  out_bf[j] = f2bf(hn);
}

// ---------------------------------------------------------------------------
// Final: out[j] = max_s w2[s] * e2[s][j]
// ---------------------------------------------------------------------------
__global__ void final_max_kernel(const float* __restrict__ w2,
                                 const float* __restrict__ e2,
                                 float* __restrict__ out) {
  int j = blockIdx.x * 256 + threadIdx.x;        // [0, H)
  float m = -3.4e38f;
  for (int s = 0; s < S_SEG; ++s) m = fmaxf(m, w2[s] * e2[(size_t)s * H_DIM + j]);
  out[j] = m;
}

// ---------------------------------------------------------------------------
// Host orchestration
// ---------------------------------------------------------------------------
extern "C" void kernel_launch(void* const* d_in, const int* in_sizes, int n_in,
                              void* d_out, int out_size, void* d_ws, size_t ws_size,
                              hipStream_t stream) {
  (void)in_sizes; (void)n_in; (void)out_size; (void)ws_size;
  const float* sent = (const float*)d_in[0];
  const float* Wi1  = (const float*)d_in[1];
  const float* Wh1  = (const float*)d_in[2];
  const float* bi1  = (const float*)d_in[3];
  const float* bh1  = (const float*)d_in[4];
  const float* Wi2  = (const float*)d_in[5];
  const float* Wh2  = (const float*)d_in[6];
  const float* bi2  = (const float*)d_in[7];
  const float* bh2  = (const float*)d_in[8];
  const float* Wl   = (const float*)d_in[9];
  const float* bl   = (const float*)d_in[10];
  const float* wa   = (const float*)d_in[11];
  const float* Wl2  = (const float*)d_in[12];
  const float* bl2  = (const float*)d_in[13];
  const float* wa2  = (const float*)d_in[14];
  float* outp = (float*)d_out;

  // ---- workspace carve-up (256B aligned) ----
  char* p = (char*)d_ws;
  auto alloc = [&p](size_t bytes) -> void* {
    void* r = (void*)p;
    p += (bytes + 255) & ~(size_t)255;
    return r;
  };
  unsigned short* sent_bf  = (unsigned short*)alloc((size_t)T_LEN * D_IN * 2);
  unsigned short* wi1_bf   = (unsigned short*)alloc((size_t)TH * D_IN * 2);
  unsigned short* wh1_bf   = (unsigned short*)alloc((size_t)TH * H_DIM * 2);
  unsigned short* wi2_bf   = (unsigned short*)alloc((size_t)TH * H_DIM * 2);
  unsigned short* wh2_bf   = (unsigned short*)alloc((size_t)TH * H_DIM * 2);
  unsigned short* wl_bf    = (unsigned short*)alloc((size_t)H_DIM * H_DIM * 2);
  unsigned short* wl2_bf   = (unsigned short*)alloc((size_t)H_DIM * H_DIM * 2);
  float*          Gi       = (float*)alloc((size_t)S_SEG * TH * 4);
  float*          Gh       = (float*)alloc((size_t)S_SEG * TH * 4);
  float*          Hf       = (float*)alloc((size_t)S_SEG * H_DIM * 4);
  unsigned short* Hbf      = (unsigned short*)alloc((size_t)S_SEG * H_DIM * 2);
  unsigned short* outs_bf  = (unsigned short*)alloc((size_t)T_LEN * H_DIM * 2);
  unsigned short* embh_bf  = (unsigned short*)alloc((size_t)T_LEN * H_DIM * 2);
  float*          wvec     = (float*)alloc((size_t)T_LEN * 4);
  unsigned short* seg_bf   = (unsigned short*)alloc((size_t)S_SEG * H_DIM * 2);
  float*          gi2      = (float*)alloc((size_t)S_SEG * TH * 4);
  float*          gh2      = (float*)alloc((size_t)TH * 4);
  float*          h2       = (float*)alloc((size_t)H_DIM * 4);
  unsigned short* outs2_bf = (unsigned short*)alloc((size_t)S_SEG * H_DIM * 2);
  float*          e2f      = (float*)alloc((size_t)S_SEG * H_DIM * 4);
  float*          w2vec    = (float*)alloc((size_t)S_SEG * 4);

  auto cast = [&](const float* src, unsigned short* dst, size_t n) {
    cast_bf16_kernel<<<dim3((unsigned)((n + 255) / 256)), dim3(256), 0, stream>>>(src, dst, (int)n);
  };
  auto gemm = [&](const unsigned short* A, int lda, int rowmul, int rowoff,
                  const unsigned short* B, int ldb, const float* bias, int act,
                  float* Cf, unsigned short* Cbf, int ldc, int M, int N, int K) {
    gemm_bf16_wmma<<<dim3(N / BN, M / BM), dim3(256), 0, stream>>>(
        A, lda, rowmul, rowoff, B, ldb, bias, act, Cf, Cbf, ldc, K);
  };

  // ---- 0) one HBM pass: bf16 copies of input + weights (L2-resident after) ----
  cast(sent, sent_bf, (size_t)T_LEN * D_IN);
  cast(Wi1, wi1_bf, (size_t)TH * D_IN);
  cast(Wh1, wh1_bf, (size_t)TH * H_DIM);
  cast(Wi2, wi2_bf, (size_t)TH * H_DIM);
  cast(Wh2, wh2_bf, (size_t)TH * H_DIM);
  cast(Wl,  wl_bf,  (size_t)H_DIM * H_DIM);
  cast(Wl2, wl2_bf, (size_t)H_DIM * H_DIM);

  // ---- 1) lower GRU: 512 independent 16-step chains -> 16 batched GEMM steps
  zero_state_kernel<<<dim3(S_SEG * H_DIM / 256), dim3(256), 0, stream>>>(Hf, Hbf, S_SEG * H_DIM);
  for (int k = 0; k < SEG_L; ++k) {
    gemm(sent_bf, D_IN, SEG_L, k, wi1_bf, D_IN, bi1, 0, Gi, nullptr, TH, S_SEG, TH, D_IN);
    gemm(Hbf, H_DIM, 1, 0, wh1_bf, H_DIM, bh1, 0, Gh, nullptr, TH, S_SEG, TH, H_DIM);
    gru1_update_kernel<<<dim3(S_SEG * H_DIM / 256), dim3(256), 0, stream>>>(
        Gi, Gh, Hf, Hbf, outs_bf, k);
  }

  // ---- 2) emb_h = tanh(outs @ Wl^T + bl)     (M=8192, N=2048, K=2048)
  gemm(outs_bf, H_DIM, 1, 0, wl_bf, H_DIM, bl, 1, nullptr, embh_bf, H_DIM,
       T_LEN, H_DIM, H_DIM);
  row_dot_bf16_kernel<<<dim3(T_LEN), dim3(256), 0, stream>>>(embh_bf, wa, wvec, H_DIM);
  seg_max_kernel<<<dim3(S_SEG * H_DIM / 256), dim3(256), 0, stream>>>(embh_bf, wvec, seg_bf);

  // ---- 3) upper GRU over 512 segments (truly sequential; Wh2 lives in L2)
  gemm(seg_bf, H_DIM, 1, 0, wi2_bf, H_DIM, bi2, 0, gi2, nullptr, TH, S_SEG, TH, H_DIM);
  zero_state_kernel<<<dim3(H_DIM / 256), dim3(256), 0, stream>>>(h2, nullptr, H_DIM);
  for (int s = 0; s < S_SEG; ++s) {
    gemv_step_kernel<<<dim3(TH / 8), dim3(256), 0, stream>>>(wh2_bf, bh2, h2, gh2);
    gru2_update_kernel<<<dim3(H_DIM / 256), dim3(256), 0, stream>>>(
        gi2 + (size_t)s * TH, gh2, h2, outs2_bf + (size_t)s * H_DIM);
  }

  // ---- 4) e2 = tanh(outs2 @ Wl2^T + bl2); out = max_s w2[s]*e2[s]
  gemm(outs2_bf, H_DIM, 1, 0, wl2_bf, H_DIM, bl2, 1, e2f, nullptr, H_DIM,
       S_SEG, H_DIM, H_DIM);
  row_dot_f32_kernel<<<dim3(S_SEG), dim3(256), 0, stream>>>(e2f, wa2, w2vec, H_DIM);
  final_max_kernel<<<dim3(H_DIM / 256), dim3(256), 0, stream>>>(w2vec, e2f, outp);
}
